// Model_39676907882627
// MI455X (gfx1250) — matrix operand order repaired and checked by reading, not compiled, not run on hardware
//
#include <hip/hip_runtime.h>

typedef _Float16 h8   __attribute__((ext_vector_type(8)));
typedef _Float16 v16h __attribute__((ext_vector_type(16)));
typedef float    v8f  __attribute__((ext_vector_type(8)));
typedef unsigned int v4u __attribute__((ext_vector_type(4)));
typedef int      v8i  __attribute__((ext_vector_type(8)));
typedef int      v4i  __attribute__((ext_vector_type(4)));

#define DIM  512
#define SEQ  8192
#define ROWS 16384  // 2 * 8192

union AFrag {
    v16h v;
    h8 h[2];
    _Float16 e[16];
};

__device__ __forceinline__ v8f wmma_f16(v16h a, v16h b, v8f c) {
    // D = A(16x32 f16) * B(32x16 f16) + C(16x16 f32)
    return __builtin_amdgcn_wmma_f32_16x16x32_f16(
        /*neg_a=*/false, a, /*neg_b=*/false, b,
        /*c_mod=*/(short)0, c, /*reuse_a=*/false, /*reuse_b=*/false);
}

// ---------------------------------------------------------------------------
// Projection: y = x @ W^T + b for (Wk,bk)->kf16, (Wq,bq)->qf16, (Wv,bv)->vT
// One 16x16 output tile per wave; K=512 contraction via 16 WMMAs.
// v is stored TRANSPOSED (vT[d][row]) so attention B-fragments are contiguous.
// ---------------------------------------------------------------------------
__global__ __launch_bounds__(256) void proj_kernel(
    const float* __restrict__ x,
    const float* __restrict__ Wk, const float* __restrict__ bk,
    const float* __restrict__ Wq, const float* __restrict__ bq,
    const float* __restrict__ Wv, const float* __restrict__ bv,
    _Float16* __restrict__ kf, _Float16* __restrict__ qf,
    _Float16* __restrict__ vT)
{
    const int wid  = (blockIdx.x * blockDim.x + threadIdx.x) >> 5;
    const int lane = threadIdx.x & 31;
    const int sel  = wid / ((ROWS / 16) * (DIM / 16));   // 0=k, 1=q, 2=v
    const int rem  = wid % ((ROWS / 16) * (DIM / 16));
    const int m0   = (rem >> 5) << 4;                    // row-tile * 16
    const int n0   = (rem & 31) << 4;                    // col-tile * 16

    const float* W    = (sel == 0) ? Wk : (sel == 1) ? Wq : Wv;
    const float* bias = (sel == 0) ? bk : (sel == 1) ? bq : bv;

    const int h   = lane >> 4;
    const int l16 = lane & 15;

    const float* xrow = x + (size_t)(m0 + l16) * DIM;
    const float* wrow = W + (size_t)(n0 + l16) * DIM;

    v8f c = {};
    for (int k0 = 0; k0 < DIM; k0 += 32) {
        AFrag a, b;
        // A (x tile, 16x32): lane row = m0+l16, K contiguous [k0+16h,+16)
        #pragma unroll
        for (int j = 0; j < 8; ++j) a.e[j]     = (_Float16)xrow[k0 + 16 * h + j];
        #pragma unroll
        for (int j = 0; j < 8; ++j) a.e[8 + j] = (_Float16)xrow[k0 + 16 * h + 8 + j];
        // B (W^T tile, 32x16): lane col = n0+l16, K contiguous [k0+16h,+16)
        #pragma unroll
        for (int j = 0; j < 16; ++j) b.e[j] = (_Float16)wrow[k0 + 16 * h + j];
        c = wmma_f16(a.v, b.v, c);
    }

    const float bb = bias[n0 + l16];
    #pragma unroll
    for (int r = 0; r < 8; ++r) c[r] += bb;

    if (sel < 2) {
        _Float16* dst = (sel == 0) ? kf : qf;
        #pragma unroll
        for (int r = 0; r < 8; ++r)
            dst[(size_t)(m0 + r + 8 * h) * DIM + n0 + l16] = (_Float16)c[r];
    } else {
        h8 o;
        #pragma unroll
        for (int r = 0; r < 8; ++r) o[r] = (_Float16)c[r];
        *(h8*)(vT + (size_t)(n0 + l16) * ROWS + m0 + 8 * h) = o;
    }
}

// ---------------------------------------------------------------------------
// Flash attention, 512 threads = 16 waves = 4 row-tile groups (64 k-rows/WG).
// Group = 4 waves; wave wq owns score K-slice [128wq,+128) (split-K, reduced
// via LDS -> no redundant score FLOPs) and output column slice [128wq,+128).
// The 64x512 f16 k-block is staged into LDS once per WG by the Tensor Data
// Mover (tensor_load_to_lds + s_wait_tensorcnt), then consumed into VGPR
// A-fragments. The 64KB k-stage is unioned with the loop-phase LDS arrays.
// ---------------------------------------------------------------------------
__global__ __launch_bounds__(512) void attn_kernel(
    const _Float16* __restrict__ kf, const _Float16* __restrict__ qf,
    const _Float16* __restrict__ vT, float* __restrict__ out)
{
    __shared__ union {
        _Float16 kTile[64 * DIM];                        // 64KB (init phase)
        struct {
            float sPart[16][512];                        // 32KB partial S
            __align__(32) _Float16 pBuf[16][512];        // 16KB P relayout
        } s;
    } sm;

    const int w    = threadIdx.x >> 5;   // wave 0..15
    const int g    = w >> 2;             // row-tile group 0..3
    const int wq   = w & 3;              // role within group
    const int lane = threadIdx.x & 31;
    const int h    = lane >> 4;
    const int l16  = lane & 15;

    const int mBase = blockIdx.x << 6;   // 64-row block (batch folded)
    const int m0    = mBase + (g << 4);  // this group's 16-row tile
    const int b     = mBase >> 13;       // batch index (whole WG same batch)
    const int tBase = b << 13;
    const int d0    = wq << 7;           // K-slice / column-slice base

    // ---- TDM: stage k block [mBase, mBase+64) x 512 f16 into LDS ----
    if (__builtin_amdgcn_readfirstlane(threadIdx.x) < 32) {  // wave 0 only
        const unsigned long long gaddr =
            (unsigned long long)(const void*)(kf + (size_t)mBase * DIM);
        const unsigned int ldsOff = (unsigned int)(unsigned long long)(void*)&sm;
        v4u g0;
        g0[0] = 1u;                       // count=1, user mode, no gather
        g0[1] = ldsOff;                   // lds_addr (bytes)
        g0[2] = (unsigned int)gaddr;      // global_addr[31:0]
        g0[3] = (unsigned int)(gaddr >> 32) | 0x80000000u; // addr[56:32] | type=2
        v8i g1;
        g1[0] = 0x00010000;               // wg_mask=0, data_size=1 (2B), no flags
        g1[1] = (int)(512u << 16);        // tensor_dim0[15:0]=512 at [63:48]
        g1[2] = 0;                        // tensor_dim0[31:16]=0, tensor_dim1[15:0]=0
        g1[3] = (int)((512u << 16) | 0x10u); // tensor_dim1[31:16]=0x10 (1M), tile_dim0=512
        g1[4] = 64;                       // tile_dim1=64, tile_dim2=0
        g1[5] = 512;                      // tensor_dim0_stride[31:0]=512
        g1[6] = 0;
        g1[7] = 0;
        v4i g2 = {0, 0, 0, 0};            // 2D: dims2/3 unused
        v4i g3 = {0, 0, 0, 0};
        v8i g4 = {0, 0, 0, 0, 0, 0, 0, 0};
        __builtin_amdgcn_tensor_load_to_lds(g0, g1, g2, g3, g4, 0);
        __builtin_amdgcn_s_wait_tensorcnt(0);
    }
    __syncthreads();

    // ---- preload k A-fragments for K in [d0, d0+128) from LDS ----
    v16h ka[4];
    {
        const _Float16* krow = sm.kTile + (size_t)((g << 4) + l16) * DIM;
        #pragma unroll
        for (int i = 0; i < 4; ++i) {
            AFrag a;
            const int base = d0 + 32 * i;
            a.h[0] = *(const h8*)(krow + base + 16 * h);
            a.h[1] = *(const h8*)(krow + base + 16 * h + 8);
            ka[i] = a.v;
        }
    }
    __syncthreads();   // kTile dead; sPart/pBuf live from here (union)

    v8f o[8];
    #pragma unroll
    for (int t = 0; t < 8; ++t) o[t] = (v8f){};
    float mstat[8], lstat[8];
    #pragma unroll
    for (int r = 0; r < 8; ++r) { mstat[r] = -1e30f; lstat[r] = 0.0f; }

    for (int t0 = 0; t0 < SEQ; t0 += 32) {
        // ---- scores: partial S over this wave's K-slice, 2 subtiles of 16
        v8f s0 = {}, s1 = {};
        #pragma unroll
        for (int i = 0; i < 4; ++i) {
            const int base = d0 + 32 * i;
            const _Float16* q0 = qf + (size_t)(tBase + t0 + l16) * DIM + base + 16 * h;
            const _Float16* q1 = q0 + (size_t)16 * DIM;
            AFrag b0, b1;
            b0.h[0] = *(const h8*)(q0);
            b0.h[1] = *(const h8*)(q0 + 8);
            b1.h[0] = *(const h8*)(q1);
            b1.h[1] = *(const h8*)(q1 + 8);
            s0 = wmma_f16(ka[i], b0.v, s0);
            s1 = wmma_f16(ka[i], b1.v, s1);
        }
        // ---- cross-wave split-K reduction (within the 4-wave group)
        #pragma unroll
        for (int r = 0; r < 8; ++r) {
            const int idx = (r + 8 * h) * 16 + l16;     // M*16 + N
            sm.s.sPart[w][idx]       = s0[r];
            sm.s.sPart[w][256 + idx] = s1[r];
        }
        __syncthreads();
        {
            const int wg0 = g << 2;
            #pragma unroll
            for (int r = 0; r < 8; ++r) {
                const int idx = (r + 8 * h) * 16 + l16;
                s0[r] = sm.s.sPart[wg0][idx] + sm.s.sPart[wg0 + 1][idx]
                      + sm.s.sPart[wg0 + 2][idx] + sm.s.sPart[wg0 + 3][idx];
                s1[r] = sm.s.sPart[wg0][256 + idx] + sm.s.sPart[wg0 + 1][256 + idx]
                      + sm.s.sPart[wg0 + 2][256 + idx] + sm.s.sPart[wg0 + 3][256 + idx];
            }
        }
        __syncthreads();

        // ---- online softmax (row M = r + 8h, reduce over 16-lane half)
        float p0[8], p1[8];
        #pragma unroll
        for (int r = 0; r < 8; ++r) {
            float rm = fmaxf(s0[r], s1[r]);
            rm = fmaxf(rm, __shfl_xor(rm, 1));
            rm = fmaxf(rm, __shfl_xor(rm, 2));
            rm = fmaxf(rm, __shfl_xor(rm, 4));
            rm = fmaxf(rm, __shfl_xor(rm, 8));
            const float mn = fmaxf(mstat[r], rm);
            const float sc = __expf(mstat[r] - mn);
            p0[r] = __expf(s0[r] - mn);
            p1[r] = __expf(s1[r] - mn);
            float rs = p0[r] + p1[r];
            rs += __shfl_xor(rs, 1);
            rs += __shfl_xor(rs, 2);
            rs += __shfl_xor(rs, 4);
            rs += __shfl_xor(rs, 8);
            lstat[r] = lstat[r] * sc + rs;
            mstat[r] = mn;
            #pragma unroll
            for (int t = 0; t < 8; ++t) o[t][r] *= sc;
        }

        // ---- relayout P: C-format f32 -> A-format f16 via per-wave LDS bounce
        #pragma unroll
        for (int r = 0; r < 8; ++r) {
            const int row = r + 8 * h;
            sm.s.pBuf[w][row * 32 + l16]      = (_Float16)p0[r];
            sm.s.pBuf[w][row * 32 + 16 + l16] = (_Float16)p1[r];
        }
        asm volatile("" ::: "memory");   // same-wave DS ops are in-order
        AFrag pa;
        {
            const _Float16* pr = &sm.s.pBuf[w][l16 * 32];
            pa.h[0] = *(const h8*)(pr + 16 * h);
            pa.h[1] = *(const h8*)(pr + 16 * h + 8);
        }

        // ---- P @ V into this wave's 8 column tiles (contiguous in vT)
        #pragma unroll
        for (int j = 0; j < 8; ++j) {
            const _Float16* vrow =
                vT + (size_t)(d0 + j * 16 + l16) * ROWS + tBase + t0 + 16 * h;
            AFrag vb;
            vb.h[0] = *(const h8*)(vrow);
            vb.h[1] = *(const h8*)(vrow + 8);
            o[j] = wmma_f16(pa.v, vb.v, o[j]);
        }
    }

    // ---- finalize: divide by softmax denom, store f32
    #pragma unroll
    for (int r = 0; r < 8; ++r) {
        const float inv = 1.0f / lstat[r];
        #pragma unroll
        for (int j = 0; j < 8; ++j) o[j][r] *= inv;
    }
    #pragma unroll
    for (int j = 0; j < 8; ++j) {
        #pragma unroll
        for (int r = 0; r < 8; ++r) {
            out[(size_t)(m0 + r + 8 * h) * DIM + d0 + j * 16 + l16] = o[j][r];
        }
    }
}

__global__ void mask_kernel(float* __restrict__ out) {
    out[(size_t)ROWS * DIM + threadIdx.x] = 0.0f;
}

extern "C" void kernel_launch(void* const* d_in, const int* in_sizes, int n_in,
                              void* d_out, int out_size, void* d_ws, size_t ws_size,
                              hipStream_t stream) {
    const float* x  = (const float*)d_in[0];
    const float* Wk = (const float*)d_in[1];
    const float* bk = (const float*)d_in[2];
    const float* Wq = (const float*)d_in[3];
    const float* bq = (const float*)d_in[4];
    const float* Wv = (const float*)d_in[5];
    const float* bv = (const float*)d_in[6];
    float* out = (float*)d_out;

    _Float16* kf = (_Float16*)d_ws;                    // 16 MB
    _Float16* qf = kf + (size_t)ROWS * DIM;            // 16 MB
    _Float16* vT = qf + (size_t)ROWS * DIM;            // 16 MB (transposed v)

    const int totalWaves = 3 * (ROWS / 16) * (DIM / 16);   // 98304
    proj_kernel<<<totalWaves / 8, 256, 0, stream>>>(x, Wk, bk, Wq, bq, Wv, bv,
                                                    kf, qf, vT);
    attn_kernel<<<ROWS / 64, 512, 0, stream>>>(kf, qf, vT, out);
    mask_kernel<<<1, 16, 0, stream>>>(out);
}
